// NeuralMemory_74071005987164
// MI455X (gfx1250) — matrix operand order repaired and checked by reading, not compiled, not run on hardware
//
#include <hip/hip_runtime.h>

typedef __bf16 bf16;
typedef __attribute__((ext_vector_type(8)))  bf16  v8bf;
typedef __attribute__((ext_vector_type(16))) bf16  v16bf;
typedef __attribute__((ext_vector_type(8)))  float v8f;

// ---------------------------------------------------------------------------
// helpers
// ---------------------------------------------------------------------------
__device__ __forceinline__ bf16 tobf(float f) {
  unsigned u = __float_as_uint(f);
  unsigned r = (u + 0x7FFFu + ((u >> 16) & 1u)) >> 16;
  unsigned short h = (unsigned short)r;
  bf16 out;
  __builtin_memcpy(&out, &h, 2);
  return out;
}
__device__ __forceinline__ float sigm_(float x) { return 1.f / (1.f + __expf(-x)); }
__device__ __forceinline__ float silu_(float x) { return x * sigm_(x); }
__device__ __forceinline__ float silubwd_(float x) {
  float s = silu_(x);
  return s + sigm_(x) * (1.f - s);
}
__device__ __forceinline__ float softplus_(float x) {
  return (x > 20.f) ? x : log1pf(__expf(x));
}

__device__ __forceinline__ v16bf cat8(v8bf lo, v8bf hi) {
  return __builtin_shufflevector(lo, hi, 0, 1, 2, 3, 4, 5, 6, 7, 8, 9, 10, 11, 12, 13, 14, 15);
}

// A fragment: row-major [16 x 32] bf16 tile (ISA 16-bit A layout):
// lane<16 -> M=lane, K = 0..7 / 16..23 ; lane>=16 -> M=lane-16, K = 8..15 / 24..31
__device__ __forceinline__ v16bf load_afrag(const bf16* __restrict__ p, int ld) {
  int lane = threadIdx.x & 31;
  const bf16* r = p + (lane & 15) * ld + ((lane >> 4) << 3);
  v8bf lo = *(const v8bf*)r;
  v8bf hi = *(const v8bf*)(r + 16);
  return cat8(lo, hi);
}
// B fragment from B^T row-major [N x K]: lane n holds K=(lane>>4)*8 .. +7 / 16+(lane>>4)*8 .. +7
__device__ __forceinline__ v16bf load_bfrag_nt(const bf16* __restrict__ p, int ld) {
  int lane = threadIdx.x & 31;
  const bf16* r = p + (lane & 15) * ld + ((lane >> 4) << 3);
  v8bf lo = *(const v8bf*)r;
  v8bf hi = *(const v8bf*)(r + 16);
  return cat8(lo, hi);
}
// f32 16x16 C/D tile: lane<16 -> N=lane, rows 0..7; lane>=16 -> rows 8..15
__device__ __forceinline__ void store_tile(float* __restrict__ p, int ld, v8f acc) {
  int lane = threadIdx.x & 31;
  int n = lane & 15;
  int mb = (lane >> 4) << 3;
#pragma unroll
  for (int i = 0; i < 8; ++i) p[(mb + i) * ld + n] = acc[i];
}

// CDNA5 async copy: global -> LDS, 16B per lane, tracked by ASYNCcnt
__device__ __forceinline__ void async_b128(unsigned lds_off, const bf16* g) {
  asm volatile("global_load_async_to_lds_b128 %0, %1, off"
               :: "v"(lds_off), "v"(g)
               : "memory");
}
__device__ __forceinline__ void wait_async0() {
#if __has_builtin(__builtin_amdgcn_s_wait_asynccnt)
  __builtin_amdgcn_s_wait_asynccnt(0);
#else
  asm volatile("s_wait_asynccnt 0x0" ::: "memory");
#endif
}

// ---------------------------------------------------------------------------
// bf16 WMMA GEMM, single orientation: C[M,N](f32) = A[M,K](rm) * B^T[N,K](rm)
// block = 8 waves (2x4), block tile 64x128, K-step 32.
// A/B slabs double-buffered in LDS via global_load_async_to_lds_b128.
// Grids must tile M,N exactly (all call sites do).
// ---------------------------------------------------------------------------
__global__ __launch_bounds__(256) void wmma_gemm_nt(
    const bf16* __restrict__ A, int lda, int sA,
    const bf16* __restrict__ B, int ldb, int sB,
    float* __restrict__ C, int ldc, int sC, int K) {
  __shared__ __align__(16) bf16 As[2][64 * 32];
  __shared__ __align__(16) bf16 Bs[2][128 * 32];
  A += blockIdx.z * sA + blockIdx.x * 64 * lda;
  B += blockIdx.z * sB + blockIdx.y * 128 * ldb;
  C += blockIdx.z * sC;
  const int t = threadIdx.x;
  const int wave = t >> 5;
  const int wr = wave >> 2, wc = wave & 3;
  const int r = t >> 2;              // staging row 0..63
  const int sg = (t & 3) << 3;       // staging column segment (elements)

  auto stage = [&](int buf, int k0) {
    __builtin_prefetch(A + r * lda + k0 + sg + 32, 0, 2);  // global_prefetch_b8
    async_b128((unsigned)(size_t)(&As[buf][r * 32 + sg]), A + r * lda + k0 + sg);
    const bf16* bg = B + r * ldb + k0 + sg;
    async_b128((unsigned)(size_t)(&Bs[buf][r * 32 + sg]), bg);
    async_b128((unsigned)(size_t)(&Bs[buf][(r + 64) * 32 + sg]), bg + 64 * ldb);
  };

  v8f c00 = {0.f, 0.f, 0.f, 0.f, 0.f, 0.f, 0.f, 0.f};
  v8f c01 = c00, c10 = c00, c11 = c00;

  const int nk = K >> 5;
  stage(0, 0);
  wait_async0();
  __syncthreads();
  for (int ki = 0; ki < nk; ++ki) {
    const int cur = ki & 1;
    if (ki + 1 < nk) stage(cur ^ 1, (ki + 1) << 5);
    const bf16* Ab = &As[cur][wr * 32 * 32];
    const bf16* Bb = &Bs[cur][wc * 32 * 32];
    v16bf a0 = load_afrag(Ab, 32);
    v16bf a1 = load_afrag(Ab + 16 * 32, 32);
    v16bf b0 = load_bfrag_nt(Bb, 32);
    v16bf b1 = load_bfrag_nt(Bb + 16 * 32, 32);
    c00 = __builtin_amdgcn_wmma_f32_16x16x32_bf16(false, a0, false, b0, (short)0, c00, false, false);
    c01 = __builtin_amdgcn_wmma_f32_16x16x32_bf16(false, a0, false, b1, (short)0, c01, false, false);
    c10 = __builtin_amdgcn_wmma_f32_16x16x32_bf16(false, a1, false, b0, (short)0, c10, false, false);
    c11 = __builtin_amdgcn_wmma_f32_16x16x32_bf16(false, a1, false, b1, (short)0, c11, false, false);
    wait_async0();     // our share of the next slab landed
    __syncthreads();   // everyone done reading cur / writing next
  }
  const int m0 = blockIdx.x * 64 + wr * 32;
  const int n0 = blockIdx.y * 128 + wc * 32;
  store_tile(C + m0 * ldc + n0, ldc, c00);
  store_tile(C + m0 * ldc + n0 + 16, ldc, c01);
  store_tile(C + (m0 + 16) * ldc + n0, ldc, c10);
  store_tile(C + (m0 + 16) * ldc + n0 + 16, ldc, c11);
}

// ---------------------------------------------------------------------------
// elementwise / scan kernels   (B=2, L=128, D=H=256 hardcoded)
// ---------------------------------------------------------------------------
__global__ void k_cvt(const float* __restrict__ s, bf16* __restrict__ d, int n) {
  int i = blockIdx.x * blockDim.x + threadIdx.x;
  if (i < n) d[i] = tobf(s[i]);
}

// transpose-convert 256x256 blocks: dst[b][c][r] = src[b][r][c]
__global__ void k_cvtT(const float* __restrict__ s, bf16* __restrict__ d) {
  int idx = blockIdx.x * 256 + threadIdx.x;
  int b = idx >> 16;
  int rr = (idx >> 8) & 255, cc = idx & 255;
  d[b * 65536 + cc * 256 + rr] = tobf(s[idx]);
}

__global__ void k_sproj(const float* __restrict__ x, const float* __restrict__ Wlr,
                        const float* __restrict__ blr, const float* __restrict__ Wm,
                        const float* __restrict__ bm, const float* __restrict__ Wd,
                        const float* __restrict__ bd, float* __restrict__ fcl,
                        float* __restrict__ fcm, float* __restrict__ fcd) {
  int t = threadIdx.x;  // 256 tokens
  const float* xr = x + t * 256;
  float s1 = 0.f, s2 = 0.f, s3 = 0.f;
  for (int d = 0; d < 256; ++d) {
    float xv = xr[d];
    s1 += xv * Wlr[d];
    s2 += xv * Wm[d];
    s3 += xv * Wd[d];
  }
  fcl[t] = s1 + blr[0] + (-4.6001660f);  // LR_SHIFT = log(expm1(0.01))
  fcm[t] = s2 + bm[0];
  fcd[t] = s3 + bd[0];
}

// decay-chain coefficients: coefA[l,j] = (Dm@M)[l,j]*lr[j], alpha, beta, e1, mom_last
__global__ void k_recur(const float* __restrict__ fcl, const float* __restrict__ fcm,
                        const float* __restrict__ fcd, float* __restrict__ alpha,
                        float* __restrict__ beta, float* __restrict__ e1c,
                        float* __restrict__ coefA, float* __restrict__ momL) {
  int b = blockIdx.x;
  int l = threadIdx.x;  // 128 threads
  __shared__ float S[128], T[128], LR[128];
  float lr = softplus_(fcl[b * 128 + l]);
  S[l] = -softplus_(-fcm[b * 128 + l]);
  T[l] = -softplus_(fcd[b * 128 + l]);
  LR[l] = lr;
  __syncthreads();
  if (l == 0) {
    float s = 0.f, t = 0.f;
    for (int i = 0; i < 128; ++i) {
      s += S[i]; S[i] = s;
      t += T[i]; T[i] = t;
    }
  }
  __syncthreads();
  float Tl = T[l], Sl = S[l];
  alpha[b * 128 + l] = __expf(Tl);
  float bsum = 0.f;
  for (int m = 0; m <= l; ++m) bsum += __expf(Tl - T[m] + S[m]);
  beta[b * 128 + l] = -bsum;
  float* row = coefA + b * 16384 + l * 128;
  for (int j = 0; j < 128; ++j) {
    float c = 0.f;
    if (j <= l) {
      float base = Tl - S[j];
      for (int m = j; m <= l; ++m) c += __expf(base - T[m] + S[m]);
      c *= LR[j];
    }
    row[j] = c;
  }
  e1c[b * 128 + l] = __expf(S[127] - Sl) * lr;
  if (l == 0) momL[b] = __expf(S[127]);
}

__global__ void k_qkvfin(const float* __restrict__ Qr, const float* __restrict__ Kr,
                         float* __restrict__ Vf, const float* __restrict__ bq,
                         const float* __restrict__ bk, const float* __restrict__ bv,
                         bf16* __restrict__ Qbf, bf16* __restrict__ Kbf,
                         float* __restrict__ Kf) {
  int idx = blockIdx.x * 256 + threadIdx.x;
  int d = idx & 255;
  Qbf[idx] = tobf(Qr[idx] + bq[d]);
  float kv = Kr[idx] + bk[d];
  Kbf[idx] = tobf(kv);
  Kf[idx] = kv;
  Vf[idx] = Vf[idx] + bv[d];  // Vf aliases raw V GEMM output
}

__global__ void k_x2(const float* __restrict__ Z1, bf16* __restrict__ X2bf) {
  int idx = blockIdx.x * 256 + threadIdx.x;
  X2bf[idx] = tobf(silu_(Z1[idx]));
}

__global__ void k_gz2(float* __restrict__ gz2f, const float* __restrict__ Vf,
                      const float* __restrict__ b2, bf16* __restrict__ gz2bf,
                      bf16* __restrict__ gz2T) {
  int idx = blockIdx.x * 256 + threadIdx.x;
  int b = idx >> 15, l = (idx >> 8) & 127, d = idx & 255;
  float g = gz2f[idx] + b2[b * 256 + d] - Vf[idx];
  gz2f[idx] = g;
  gz2bf[idx] = tobf(g);
  gz2T[b * 32768 + d * 128 + l] = tobf(g);
}

__global__ void k_gz1(float* __restrict__ gz1f, const float* __restrict__ Z1,
                      bf16* __restrict__ gz1T) {
  int idx = blockIdx.x * 256 + threadIdx.x;
  int b = idx >> 15, l = (idx >> 8) & 127, h = idx & 255;
  float g = gz1f[idx] * silubwd_(Z1[idx]);
  gz1f[idx] = g;
  gz1T[b * 32768 + h * 128 + l] = tobf(g);
}

__global__ void k_emake(const float* __restrict__ coefA, const float* __restrict__ Sm,
                        bf16* __restrict__ E) {
  int idx = blockIdx.x * 256 + threadIdx.x;  // 32768
  E[idx] = tobf(coefA[idx] * (1.f + Sm[idx]));
}

__global__ void k_zq1fin(const float* __restrict__ Zq1r, const float* __restrict__ G1,
                         const float* __restrict__ Gm1, const float* __restrict__ b1,
                         const float* __restrict__ mb1, const float* __restrict__ alpha,
                         const float* __restrict__ beta, bf16* __restrict__ Xq2bf) {
  int idx = blockIdx.x * 256 + threadIdx.x;
  int b = idx >> 15, l = (idx >> 8) & 127, h = idx & 255;
  int t = b * 128 + l;
  float z = Zq1r[idx] + alpha[t] * (G1[idx] + b1[b * 256 + h]) +
            beta[t] * (Gm1[idx] + mb1[b * 256 + h]);
  Xq2bf[idx] = tobf(silu_(z));
}

__global__ void k_zq2fin(const float* __restrict__ Zq2r, const float* __restrict__ G2,
                         const float* __restrict__ Gm2, const float* __restrict__ b2,
                         const float* __restrict__ mb2, const float* __restrict__ alpha,
                         const float* __restrict__ beta, float* __restrict__ out) {
  int idx = blockIdx.x * 256 + threadIdx.x;
  int b = idx >> 15, l = (idx >> 8) & 127, d = idx & 255;
  int t = b * 128 + l;
  out[idx] = Zq2r[idx] + alpha[t] * (G2[idx] + b2[b * 256 + d]) +
             beta[t] * (Gm2[idx] + mb2[b * 256 + d]);
}

// coefficient-scaled key / activation copies, written TRANSPOSED: [B, D|H, L]
__global__ void k_ksmake(const float* __restrict__ coefA, const float* __restrict__ e1c,
                         const float* __restrict__ Kf, const float* __restrict__ Z1,
                         bf16* __restrict__ Ks1T, bf16* __restrict__ Ks2T,
                         bf16* __restrict__ X2s1T, bf16* __restrict__ X2s2T) {
  int idx = blockIdx.x * 256 + threadIdx.x;
  int b = idx >> 15, l = (idx >> 8) & 127, c = idx & 255;
  float c1 = coefA[b * 16384 + 127 * 128 + l];
  float c2 = e1c[b * 128 + l];
  int to = b * 32768 + c * 128 + l;
  if (blockIdx.y == 0) {
    float kv = Kf[idx];
    Ks1T[to] = tobf(c1 * kv);
    Ks2T[to] = tobf(c2 * kv);
  } else {
    float xv = silu_(Z1[idx]);
    X2s1T[to] = tobf(c1 * xv);
    X2s2T[to] = tobf(c2 * xv);
  }
}

__global__ void k_wlast(const float* __restrict__ W1, const float* __restrict__ W2,
                        const float* __restrict__ mW1, const float* __restrict__ mW2,
                        const float* __restrict__ alpha, const float* __restrict__ beta,
                        const float* __restrict__ momL, float* __restrict__ out) {
  int idx = blockIdx.x * 256 + threadIdx.x;  // 0..131071
  int b = idx >> 16;
  float aL = alpha[b * 128 + 127], bL = beta[b * 128 + 127], mL = momL[b];
  switch (blockIdx.y) {
    case 0: out[65536 + idx] += aL * W1[idx] + bL * mW1[idx]; break;
    case 1: out[197120 + idx] += aL * W2[idx] + bL * mW2[idx]; break;
    case 2: out[328704 + idx] -= mL * mW1[idx]; break;
    default: out[460288 + idx] -= mL * mW2[idx]; break;
  }
}

__global__ void k_bvec(const float* __restrict__ gz1f, const float* __restrict__ gz2f,
                       const float* __restrict__ coefA, const float* __restrict__ e1c,
                       const float* __restrict__ alpha, const float* __restrict__ beta,
                       const float* __restrict__ momL, const float* __restrict__ b1,
                       const float* __restrict__ mb1, const float* __restrict__ b2,
                       const float* __restrict__ mb2, float* __restrict__ out) {
  int t = blockIdx.x * 256 + threadIdx.x;  // 0..511
  int path = blockIdx.y;
  int b = t >> 8, n = t & 255;
  const float* g = (path == 0 ? gz1f : gz2f) + b * 32768 + n;
  const float* cL = coefA + b * 16384 + 127 * 128;
  const float* e1 = e1c + b * 128;
  float s1 = 0.f, s2 = 0.f;
  for (int j = 0; j < 128; ++j) {
    float gv = g[j * 256];
    s1 += cL[j] * gv;
    s2 += e1[j] * gv;
  }
  float aL = alpha[b * 128 + 127], bL = beta[b * 128 + 127], mL = momL[b];
  if (path == 0) {
    out[196608 + b * 256 + n] = s1 + aL * b1[b * 256 + n] + bL * mb1[b * 256 + n];
    out[459776 + b * 256 + n] = s2 - mL * mb1[b * 256 + n];
  } else {
    out[328192 + b * 256 + n] = s1 + aL * b2[b * 256 + n] + bL * mb2[b * 256 + n];
    out[591360 + b * 256 + n] = s2 - mL * mb2[b * 256 + n];
  }
}

// ---------------------------------------------------------------------------
// host launcher
// ---------------------------------------------------------------------------
extern "C" void kernel_launch(void* const* d_in, const int* in_sizes, int n_in,
                              void* d_out, int out_size, void* d_ws, size_t ws_size,
                              hipStream_t stream) {
  (void)in_sizes; (void)n_in; (void)out_size; (void)ws_size;
  const float* x   = (const float*)d_in[0];
  const float* Wq  = (const float*)d_in[1];
  const float* bq  = (const float*)d_in[2];
  const float* Wk  = (const float*)d_in[3];
  const float* bk  = (const float*)d_in[4];
  const float* Wv  = (const float*)d_in[5];
  const float* bv  = (const float*)d_in[6];
  const float* Wlr = (const float*)d_in[7];
  const float* blr = (const float*)d_in[8];
  const float* Wm  = (const float*)d_in[9];
  const float* bm  = (const float*)d_in[10];
  const float* Wd  = (const float*)d_in[11];
  const float* bd  = (const float*)d_in[12];
  const float* W1  = (const float*)d_in[13];
  const float* b1  = (const float*)d_in[14];
  const float* W2  = (const float*)d_in[15];
  const float* b2  = (const float*)d_in[16];
  const float* mW1 = (const float*)d_in[17];
  const float* mb1 = (const float*)d_in[18];
  const float* mW2 = (const float*)d_in[19];
  const float* mb2 = (const float*)d_in[20];
  float* out = (float*)d_out;

  char* w = (char*)d_ws;
  auto alloc = [&](size_t bytes) {
    char* p = w;
    w += (bytes + 255) & ~(size_t)255;
    return p;
  };
  // bf16 buffers
  bf16* xbf   = (bf16*)alloc(65536 * 2);
  bf16* WqT   = (bf16*)alloc(65536 * 2);
  bf16* WkT   = (bf16*)alloc(65536 * 2);
  bf16* WvT   = (bf16*)alloc(65536 * 2);
  bf16* W1bf  = (bf16*)alloc(131072 * 2);
  bf16* W2bf  = (bf16*)alloc(131072 * 2);
  bf16* W2T   = (bf16*)alloc(131072 * 2);
  bf16* mW1bf = (bf16*)alloc(131072 * 2);
  bf16* mW2bf = (bf16*)alloc(131072 * 2);
  bf16* Qbf   = (bf16*)alloc(65536 * 2);
  bf16* Kbf   = (bf16*)alloc(65536 * 2);
  bf16* X2bf  = (bf16*)alloc(65536 * 2);
  bf16* gz2bf = (bf16*)alloc(65536 * 2);
  bf16* gz1T  = (bf16*)alloc(65536 * 2);
  bf16* gz2T  = (bf16*)alloc(65536 * 2);
  bf16* E1    = (bf16*)alloc(32768 * 2);
  bf16* E2    = (bf16*)alloc(32768 * 2);
  bf16* Xq2bf = (bf16*)alloc(65536 * 2);
  bf16* Ks1T  = (bf16*)alloc(65536 * 2);
  bf16* Ks2T  = (bf16*)alloc(65536 * 2);
  bf16* X2s1T = (bf16*)alloc(65536 * 2);
  bf16* X2s2T = (bf16*)alloc(65536 * 2);
  // f32 buffers
  float* Qr    = (float*)alloc(65536 * 4);
  float* Kr    = (float*)alloc(65536 * 4);
  float* Vf    = (float*)alloc(65536 * 4);  // raw V -> V+bias in place
  float* Kf    = (float*)alloc(65536 * 4);
  float* Z1    = (float*)alloc(65536 * 4);
  float* gz2f  = (float*)alloc(65536 * 4);  // raw Z2 -> grad_Z2 in place
  float* gz1f  = (float*)alloc(65536 * 4);  // raw grad_X2 -> grad_Z1 in place
  float* Sm    = (float*)alloc(32768 * 4);
  float* S2    = (float*)alloc(32768 * 4);
  float* G1    = (float*)alloc(65536 * 4);
  float* Gm1   = (float*)alloc(65536 * 4);
  float* G2    = (float*)alloc(65536 * 4);
  float* Gm2   = (float*)alloc(65536 * 4);
  float* Zq1r  = (float*)alloc(65536 * 4);
  float* Zq2r  = (float*)alloc(65536 * 4);
  float* coefA = (float*)alloc(32768 * 4);
  float* fcl   = (float*)alloc(256 * 4);
  float* fcm   = (float*)alloc(256 * 4);
  float* fcd   = (float*)alloc(256 * 4);
  float* alpha = (float*)alloc(256 * 4);
  float* beta  = (float*)alloc(256 * 4);
  float* e1c   = (float*)alloc(256 * 4);
  float* momL  = (float*)alloc(2 * 4);

  // C[M,N] = A[M,K] * B^T[N,K];  grids tile exactly
  auto gemm = [&](const void* Ap, int lda, int sA, const void* Bp, int ldb, int sB,
                  float* Cp, int ldc, int sC, int M, int N, int K, int nb) {
    dim3 g((unsigned)(M / 64), (unsigned)(N / 128), (unsigned)nb);
    wmma_gemm_nt<<<g, 256, 0, stream>>>((const bf16*)Ap, lda, sA, (const bf16*)Bp, ldb,
                                        sB, Cp, ldc, sC, K);
  };

  // ---- conversions (straight + transposed) ----
  k_cvt<<<256, 256, 0, stream>>>(x, xbf, 65536);
  k_cvtT<<<256, 256, 0, stream>>>(Wq, WqT);
  k_cvtT<<<256, 256, 0, stream>>>(Wk, WkT);
  k_cvtT<<<256, 256, 0, stream>>>(Wv, WvT);
  k_cvt<<<512, 256, 0, stream>>>(W1, W1bf, 131072);
  k_cvt<<<512, 256, 0, stream>>>(W2, W2bf, 131072);
  k_cvtT<<<512, 256, 0, stream>>>(W2, W2T);
  k_cvt<<<512, 256, 0, stream>>>(mW1, mW1bf, 131072);
  k_cvt<<<512, 256, 0, stream>>>(mW2, mW2bf, 131072);

  // ---- scalar projections + decay-chain coefficients ----
  k_sproj<<<1, 256, 0, stream>>>(x, Wlr, blr, Wm, bm, Wd, bd, fcl, fcm, fcd);
  k_recur<<<2, 128, 0, stream>>>(fcl, fcm, fcd, alpha, beta, e1c, coefA, momL);

  // ---- Q/K/V projections ----
  gemm(xbf, 256, 0, WqT, 256, 0, Qr, 256, 0, 256, 256, 256, 1);
  gemm(xbf, 256, 0, WkT, 256, 0, Kr, 256, 0, 256, 256, 256, 1);
  gemm(xbf, 256, 0, WvT, 256, 0, Vf, 256, 0, 256, 256, 256, 1);
  k_qkvfin<<<256, 256, 0, stream>>>(Qr, Kr, Vf, bq, bk, bv, Qbf, Kbf, Kf);

  // ---- fast-weight forward + per-token gradients ----
  gemm(Kbf, 256, 32768, W1bf, 256, 65536, Z1, 256, 32768, 128, 256, 256, 2);
  k_x2<<<256, 256, 0, stream>>>(Z1, X2bf);
  gemm(X2bf, 256, 32768, W2bf, 256, 65536, gz2f, 256, 32768, 128, 256, 256, 2);
  k_gz2<<<256, 256, 0, stream>>>(gz2f, Vf, b2, gz2bf, gz2T);
  gemm(gz2bf, 256, 32768, W2T, 256, 65536, gz1f, 256, 32768, 128, 256, 256, 2);
  k_gz1<<<256, 256, 0, stream>>>(gz1f, Z1, gz1T);

  // ---- query pass, layer 1 (linear-attention form) ----
  gemm(Qbf, 256, 32768, Kbf, 256, 32768, Sm, 128, 16384, 128, 128, 256, 2);
  gemm(Qbf, 256, 32768, W1bf, 256, 65536, G1, 256, 32768, 128, 256, 256, 2);
  gemm(Qbf, 256, 32768, mW1bf, 256, 65536, Gm1, 256, 32768, 128, 256, 256, 2);
  k_emake<<<128, 256, 0, stream>>>(coefA, Sm, E1);
  gemm(E1, 128, 16384, gz1T, 128, 32768, Zq1r, 256, 32768, 128, 256, 128, 2);
  k_zq1fin<<<256, 256, 0, stream>>>(Zq1r, G1, Gm1, b1, mb1, alpha, beta, Xq2bf);

  // ---- query pass, layer 2 ----
  gemm(Xq2bf, 256, 32768, X2bf, 256, 32768, S2, 128, 16384, 128, 128, 256, 2);
  gemm(Xq2bf, 256, 32768, W2bf, 256, 65536, G2, 256, 32768, 128, 256, 256, 2);
  gemm(Xq2bf, 256, 32768, mW2bf, 256, 65536, Gm2, 256, 32768, 128, 256, 256, 2);
  k_emake<<<128, 256, 0, stream>>>(coefA, S2, E2);
  gemm(E2, 128, 16384, gz2T, 128, 32768, Zq2r, 256, 32768, 128, 256, 128, 2);
  k_zq2fin<<<256, 256, 0, stream>>>(Zq2r, G2, Gm2, b2, mb2, alpha, beta, out);

  // ---- last-token weights / momenta ----
  k_ksmake<<<dim3(256, 2), 256, 0, stream>>>(coefA, e1c, Kf, Z1, Ks1T, Ks2T, X2s1T, X2s2T);
  gemm(gz1T, 128, 32768, Ks1T, 128, 32768, out + 65536, 256, 65536, 256, 256, 128, 2);
  gemm(gz1T, 128, 32768, Ks2T, 128, 32768, out + 328704, 256, 65536, 256, 256, 128, 2);
  gemm(gz2T, 128, 32768, X2s1T, 128, 32768, out + 197120, 256, 65536, 256, 256, 128, 2);
  gemm(gz2T, 128, 32768, X2s2T, 128, 32768, out + 460288, 256, 65536, 256, 256, 128, 2);
  k_wlast<<<dim3(512, 4), 256, 0, stream>>>(W1, W2, mW1, mW2, alpha, beta, momL, out);
  k_bvec<<<dim3(2, 2), 256, 0, stream>>>(gz1f, gz2f, coefA, e1c, alpha, beta, momL,
                                         b1, mb1, b2, mb2, out);
}
